// IPGAT_90486370992456
// MI455X (gfx1250) — matrix operand order repaired and checked by reading, not compiled, not run on hardware
//
#include <hip/hip_runtime.h>
#include <cstdint>
#include <cstring>

// ---------------- constants ----------------
#define NN     4096
#define EMB    256
#define ATT    128
#define NHEAD  3
#define NPAIRS 131072
#define FEAT   (NHEAD * ATT)      // 384
#define NEGV   (-9.0e15f)
#define ALPHA_LRELU 0.2f
#define LROW   20                 // padded LDS row stride in dwords (16B-aligned, conflict-free)

// ---------------- WMMA types ----------------
typedef __attribute__((ext_vector_type(16))) __bf16 v16bf;
typedef __attribute__((ext_vector_type(8)))  float  v8f;

union BF16x16 { v16bf v; __bf16 h[16]; uint32_t u[8]; };

static __device__ __forceinline__ float lrelu(float x) {
  return x > 0.f ? x : ALPHA_LRELU * x;
}
static __device__ __forceinline__ float gelu_exact(float x) {
  return 0.5f * x * (1.f + erff(x * 0.70710678118654752f));
}

// A-fragment K-offset (contiguous K order, matching B):
// lanes 0-15 (half=0): v0..7 -> K=2v
// lanes16-31 (half=1): v0..7 -> K=16+2v
static __device__ __forceinline__ int a_koff(int v, int half) {
  return 16 * half + 2 * v;
}

// Async byte-copy of a 32(K) x 128(N) bf16 tile (K contiguous per row) into padded LDS.
// 512 16-byte chunks spread over 128 threads; all addresses 16B-aligned.
static __device__ __forceinline__ void async_tile_load(
    const __bf16* base, size_t rstride, int jb, uint32_t ldsb, int tid) {
  #pragma unroll
  for (int c = 0; c < 4; ++c) {
    int idx4 = c * 128 + tid;                          // 0..511
    int n = idx4 >> 2, q = idx4 & 3;
    uint64_t ga = (uint64_t)(uintptr_t)(base + (size_t)n * rstride + jb + 8 * q);
    uint32_t la = ldsb + 4u * (uint32_t)(n * LROW + 4 * q);
    asm volatile("global_load_async_to_lds_b128 %0, %1, off"
                 :: "v"(la), "v"(ga) : "memory");
  }
}
static __device__ __forceinline__ void async_wait0() {
  asm volatile("s_wait_asynccnt 0" ::: "memory");
}

// ---------------- kernel 1: pack adjacency to bitmask ----------------
__global__ void pack_adj_k(const int* __restrict__ adj, uint32_t* __restrict__ bits) {
  int w = blockIdx.x * blockDim.x + threadIdx.x;      // 4096*128 words
  int i = w >> 7, c = (w & 127) << 5;
  const int* row = adj + (size_t)i * NN + c;
  uint32_t m = 0;
  #pragma unroll
  for (int b = 0; b < 32; ++b) m |= (row[b] > 0 ? 1u : 0u) << b;
  bits[w] = m;
}

// -------- kernel 1b: transpose+convert weights: in[b][K][Nc] f32 -> out[b][Nc][K] bf16 ----
__global__ void wtrans_k(const float* __restrict__ in, __bf16* __restrict__ out,
                         int K, int Nc, int total) {
  int t = blockIdx.x * blockDim.x + threadIdx.x;
  if (t >= total) return;
  int k = t % K;
  int n = (t / K) % Nc;
  int b = t / (K * Nc);
  out[((size_t)b * Nc + n) * K + k] = (__bf16)in[((size_t)b * K + k) * Nc + n];
}

// ---------------- kernel 2: Wh = X @ W  (per head), output WhT bf16 [H][ATT][N] --------
__global__ __launch_bounds__(128) void node_gemm_k(
    const float* __restrict__ X, int K,
    const __bf16* __restrict__ WT,             // [H][ATT][K] bf16 (pre-transposed)
    __bf16* __restrict__ WhT) {                // [H][ATT][N] bf16
  __shared__ uint32_t lb[2][128 * LROW];
  int tid = threadIdx.x;
  int lane = tid & 31;
  int gw = blockIdx.x * 4 + (tid >> 5);                // 768 waves
  int h = gw >> 8, tile = gw & 255;
  int nb = tile * 16;
  int mr = lane & 15, half = lane >> 4;
  const __bf16* Bsrc = WT + (size_t)h * ATT * K;
  const float* Xr = X + (size_t)(nb + mr) * K;
  uint32_t ldsb[2] = { (uint32_t)(uintptr_t)&lb[0][0], (uint32_t)(uintptr_t)&lb[1][0] };

  v8f acc[8];
  #pragma unroll
  for (int f = 0; f < 8; ++f)
    #pragma unroll
    for (int r = 0; r < 8; ++r) acc[f][r] = 0.f;

  int ksteps = K >> 5;
  async_tile_load(Bsrc, K, 0, ldsb[0], tid);
  for (int t = 0; t < ksteps; ++t) {
    int kb = t << 5;
    async_wait0();
    __syncthreads();
    if (t + 1 < ksteps) async_tile_load(Bsrc, K, (t + 1) << 5, ldsb[(t + 1) & 1], tid);
    const uint32_t* cur = lb[t & 1];

    BF16x16 af;
    #pragma unroll
    for (int v = 0; v < 8; ++v) {
      int kd = a_koff(v, half);
      float2 x2 = *reinterpret_cast<const float2*>(Xr + kb + kd);
      af.h[2 * v]     = (__bf16)x2.x;
      af.h[2 * v + 1] = (__bf16)x2.y;
    }
    #pragma unroll
    for (int f = 0; f < 8; ++f) {
      BF16x16 bfr;
      int n = f * 16 + mr;
      #pragma unroll
      for (int v = 0; v < 8; ++v) bfr.u[v] = cur[n * LROW + 8 * half + v];
      acc[f] = __builtin_amdgcn_wmma_f32_16x16x32_bf16(
          false, af.v, false, bfr.v, (short)0, acc[f], false, false);
    }
  }
  // C layout: lane holds (M = r + 8*half, N = f*16 + mr)
  #pragma unroll
  for (int f = 0; f < 8; ++f) {
    int n = f * 16 + mr;
    __bf16* dst = WhT + ((size_t)(h * ATT + n)) * NN + nb;
    #pragma unroll
    for (int r = 0; r < 8; ++r) dst[r + 8 * half] = (__bf16)acc[f][r];
  }
}

// ---------------- kernel 3: el/er per node per head ----------------
__global__ void el_er_k(const __bf16* __restrict__ WhT,
                        const float* __restrict__ av,    // [H][256]
                        float* __restrict__ el, float* __restrict__ er) {
  int t = blockIdx.x * blockDim.x + threadIdx.x;
  if (t >= NHEAD * NN) return;
  int h = t >> 12, i = t & (NN - 1);
  const float* avh = av + h * 2 * ATT;
  float sl = 0.f, sr = 0.f;
  for (int f = 0; f < ATT; ++f) {
    float w = (float)WhT[((size_t)(h * ATT + f)) * NN + i];
    sl += w * avh[f];
    sr += w * avh[ATT + f];
  }
  el[t] = sl; er[t] = sr;
}

// ---------------- kernel 4: per-row softmax stats (max, sum) ----------------
__global__ __launch_bounds__(256) void row_stats_k(
    const float* __restrict__ el, const float* __restrict__ er,
    const uint32_t* __restrict__ bits,
    float* __restrict__ mrow, float* __restrict__ srow) {
  int lane = threadIdx.x & 31;
  int gw = blockIdx.x * 8 + (threadIdx.x >> 5);        // H*N waves
  int h = gw >> 12, i = gw & (NN - 1);
  float eli = el[gw];
  const float* erh = er + (h << 12);
  const uint32_t* br = bits + (size_t)i * 128;
  float mx = NEGV;
  for (int t = 0; t < 128; ++t) {
    uint32_t aw = br[t];
    float e = ((aw >> lane) & 1u) ? lrelu(eli + erh[t * 32 + lane]) : NEGV;
    mx = fmaxf(mx, e);
  }
  #pragma unroll
  for (int off = 16; off; off >>= 1) mx = fmaxf(mx, __shfl_xor(mx, off, 32));
  float sum = 0.f;
  for (int t = 0; t < 128; ++t) {
    uint32_t aw = br[t];
    float e = ((aw >> lane) & 1u) ? lrelu(eli + erh[t * 32 + lane]) : NEGV;
    sum += __expf(e - mx);
  }
  #pragma unroll
  for (int off = 16; off; off >>= 1) sum += __shfl_xor(sum, off, 32);
  if (lane == 0) { mrow[gw] = mx; srow[gw] = sum; }
}

// ---------------- kernel 5: out = elu(softmax(P) @ Wh), P generated on the fly -------
__global__ __launch_bounds__(128) void agg_k(
    const __bf16* __restrict__ WhT,
    const float* __restrict__ el, const float* __restrict__ er,
    const float* __restrict__ mrow, const float* __restrict__ srow,
    const uint32_t* __restrict__ bits,
    float* __restrict__ out_at) {                      // [N][FEAT]
  __shared__ uint32_t lb[2][128 * LROW];
  int tid = threadIdx.x;
  int lane = tid & 31;
  int gw = blockIdx.x * 4 + (tid >> 5);
  int h = gw >> 8, tile = gw & 255;
  int mr = lane & 15, half = lane >> 4;
  int i = tile * 16 + mr;
  int hi = (h << 12) + i;
  float eli = el[hi], mi = mrow[hi], inv_s = 1.f / srow[hi];
  const float* erh = er + (h << 12);
  const uint32_t* br = bits + (size_t)i * 128;
  const __bf16* Wh = WhT + ((size_t)h * ATT) * NN;
  uint32_t ldsb[2] = { (uint32_t)(uintptr_t)&lb[0][0], (uint32_t)(uintptr_t)&lb[1][0] };

  v8f acc[8];
  #pragma unroll
  for (int f = 0; f < 8; ++f)
    #pragma unroll
    for (int r = 0; r < 8; ++r) acc[f][r] = 0.f;

  async_tile_load(Wh, NN, 0, ldsb[0], tid);
  for (int t = 0; t < 128; ++t) {                      // K-loop over 4096 neighbors
    int jb = t << 5;
    async_wait0();
    __syncthreads();
    if (t + 1 < 128) async_tile_load(Wh, NN, (t + 1) << 5, ldsb[(t + 1) & 1], tid);
    const uint32_t* cur = lb[t & 1];

    uint32_t aw = br[t];
    BF16x16 af;
    #pragma unroll
    for (int v = 0; v < 8; ++v) {
      int kd = a_koff(v, half);
      float2 e2 = *reinterpret_cast<const float2*>(erh + jb + kd);
      float e0 = ((aw >> kd) & 1u)       ? lrelu(eli + e2.x) : NEGV;
      float e1 = ((aw >> (kd + 1)) & 1u) ? lrelu(eli + e2.y) : NEGV;
      af.h[2 * v]     = (__bf16)(__expf(e0 - mi) * inv_s);
      af.h[2 * v + 1] = (__bf16)(__expf(e1 - mi) * inv_s);
    }
    #pragma unroll
    for (int f = 0; f < 8; ++f) {
      BF16x16 bfr;
      int n = f * 16 + mr;
      #pragma unroll
      for (int v = 0; v < 8; ++v) bfr.u[v] = cur[n * LROW + 8 * half + v];
      acc[f] = __builtin_amdgcn_wmma_f32_16x16x32_bf16(
          false, af.v, false, bfr.v, (short)0, acc[f], false, false);
    }
  }
  #pragma unroll
  for (int f = 0; f < 8; ++f) {
    int n = f * 16 + mr;
    #pragma unroll
    for (int r = 0; r < 8; ++r) {
      int node = tile * 16 + r + 8 * half;
      float x = acc[f][r];
      out_at[(size_t)node * FEAT + h * ATT + n] = x > 0.f ? x : __expf(x) - 1.f; // ELU
    }
  }
}

// ---------------- kernel 6: atten = at1*wa + at2*wb ----------------
__global__ void mix_k(const float* __restrict__ at1, const float* __restrict__ at2,
                      const float* __restrict__ amix, const float* __restrict__ bmix,
                      float* __restrict__ atten) {
  int t = blockIdx.x * blockDim.x + threadIdx.x;
  if (t >= NN * FEAT) return;
  float a = amix[0], b = bmix[0], d = a + b;
  atten[t] = at1[t] * (a / d) + at2[t] * (b / d);
}

// ------ kernel 7: fused pair scorer: (atten[a]*atten[b]) @ linW -> gelu -> dot(outW) ->
//        + gelu(emb_b[a]+emb_b[b])*outW[128] + out_b -> sigmoid ------
__global__ __launch_bounds__(128) void pair_k(
    const float* __restrict__ atten,
    const int* __restrict__ aix, const int* __restrict__ bix,
    const __bf16* __restrict__ linWT, // [ATT][FEAT] bf16 (pre-transposed)
    const float* __restrict__ linb,   // [ATT]
    const float* __restrict__ embb,   // [N]
    const float* __restrict__ outW,   // [ATT+1]
    const float* __restrict__ outb,   // [1]
    float* __restrict__ out) {        // [NPAIRS]
  __shared__ uint32_t lb[2][128 * LROW];
  int tid = threadIdx.x;
  int lane = tid & 31;
  int tile = blockIdx.x * 4 + (tid >> 5);              // 8192 tiles of 16 pairs
  int mr = lane & 15, half = lane >> 4;
  int p = tile * 16 + mr;
  const float* ra = atten + (size_t)aix[p] * FEAT;
  const float* rb = atten + (size_t)bix[p] * FEAT;
  uint32_t ldsb[2] = { (uint32_t)(uintptr_t)&lb[0][0], (uint32_t)(uintptr_t)&lb[1][0] };

  v8f acc[8];
  #pragma unroll
  for (int f = 0; f < 8; ++f)
    #pragma unroll
    for (int r = 0; r < 8; ++r) acc[f][r] = 0.f;

  const int KST = FEAT / 32;                           // 12 K-steps
  async_tile_load(linWT, FEAT, 0, ldsb[0], tid);
  for (int t = 0; t < KST; ++t) {
    int kb = t << 5;
    async_wait0();
    __syncthreads();
    if (t + 1 < KST) async_tile_load(linWT, FEAT, (t + 1) << 5, ldsb[(t + 1) & 1], tid);
    const uint32_t* cur = lb[t & 1];

    BF16x16 af;
    #pragma unroll
    for (int v = 0; v < 8; ++v) {
      int kd = a_koff(v, half);
      float2 xa = *reinterpret_cast<const float2*>(ra + kb + kd);
      float2 xb = *reinterpret_cast<const float2*>(rb + kb + kd);
      af.h[2 * v]     = (__bf16)(xa.x * xb.x);
      af.h[2 * v + 1] = (__bf16)(xa.y * xb.y);
    }
    #pragma unroll
    for (int f = 0; f < 8; ++f) {
      BF16x16 bfr;
      int n = f * 16 + mr;
      #pragma unroll
      for (int v = 0; v < 8; ++v) bfr.u[v] = cur[n * LROW + 8 * half + v];
      acc[f] = __builtin_amdgcn_wmma_f32_16x16x32_bf16(
          false, af.v, false, bfr.v, (short)0, acc[f], false, false);
    }
  }

  // epilogue: gelu + dot with outW over N=128, reduced across the 16 lanes of each half
  float pr[8];
  #pragma unroll
  for (int r = 0; r < 8; ++r) pr[r] = 0.f;
  #pragma unroll
  for (int f = 0; f < 8; ++f) {
    int n = f * 16 + mr;
    float bn = linb[n], wn = outW[n];
    #pragma unroll
    for (int r = 0; r < 8; ++r) pr[r] += gelu_exact(acc[f][r] + bn) * wn;
  }
  #pragma unroll
  for (int r = 0; r < 8; ++r) {
    #pragma unroll
    for (int off = 8; off; off >>= 1) pr[r] += __shfl_xor(pr[r], off, 16);
  }
  if (mr == 0) {
    float w128 = outW[ATT], ob = outb[0];
    #pragma unroll
    for (int r = 0; r < 8; ++r) {
      int pp = tile * 16 + r + 8 * half;
      float ba = embb[aix[pp]] + embb[bix[pp]];
      float z = pr[r] + gelu_exact(ba) * w128 + ob;
      out[pp] = 1.f / (1.f + __expf(-z));
    }
  }
}

// ---------------- host-side orchestration ----------------
extern "C" void kernel_launch(void* const* d_in, const int* in_sizes, int n_in,
                              void* d_out, int out_size, void* d_ws, size_t ws_size,
                              hipStream_t stream) {
  (void)in_sizes; (void)n_in; (void)out_size; (void)ws_size;
  const int*   a_idx = (const int*)  d_in[0];
  const int*   b_idx = (const int*)  d_in[1];
  const int*   adj   = (const int*)  d_in[2];
  const float* emb   = (const float*)d_in[3];
  const float* emb_b = (const float*)d_in[4];
  const float* W1    = (const float*)d_in[5];
  const float* av1   = (const float*)d_in[6];
  const float* W2    = (const float*)d_in[7];
  const float* av2   = (const float*)d_in[8];
  const float* a_mix = (const float*)d_in[9];
  const float* b_mix = (const float*)d_in[10];
  const float* lin_W = (const float*)d_in[11];
  const float* lin_b = (const float*)d_in[12];
  const float* out_W = (const float*)d_in[13];
  const float* out_b = (const float*)d_in[14];
  float* out = (float*)d_out;

  // workspace carve-out (all offsets 256B aligned)
  uint8_t* ws = (uint8_t*)d_ws;
  size_t o = 0;
  auto take = [&](size_t bytes) { uint8_t* p = ws + o; o += (bytes + 255) & ~(size_t)255; return p; };
  uint32_t* adjbits = (uint32_t*)take((size_t)NN * 128 * 4);                 // 2 MB
  __bf16*   WhT     = (__bf16*)  take((size_t)NHEAD * ATT * NN * 2);        // 3 MB
  __bf16*   W1T     = (__bf16*)  take((size_t)NHEAD * ATT * EMB * 2);
  __bf16*   W2T     = (__bf16*)  take((size_t)NHEAD * ATT * FEAT * 2);
  __bf16*   linWT   = (__bf16*)  take((size_t)ATT * FEAT * 2);
  float*    el      = (float*)   take((size_t)NHEAD * NN * 4);
  float*    er      = (float*)   take((size_t)NHEAD * NN * 4);
  float*    mrow    = (float*)   take((size_t)NHEAD * NN * 4);
  float*    srow    = (float*)   take((size_t)NHEAD * NN * 4);
  float*    at1     = (float*)   take((size_t)NN * FEAT * 4);               // 6 MB
  float*    at2     = (float*)   take((size_t)NN * FEAT * 4);               // 6 MB
  float*    atten   = (float*)   take((size_t)NN * FEAT * 4);               // 6 MB

  // 0) pack adjacency into bitmask; pre-transpose weights to bf16 K-major
  pack_adj_k<<<(NN * 128) / 256, 256, 0, stream>>>(adj, adjbits);
  { int tot = NHEAD * EMB * ATT;
    wtrans_k<<<(tot + 255) / 256, 256, 0, stream>>>(W1, W1T, EMB, ATT, tot); }
  { int tot = NHEAD * FEAT * ATT;
    wtrans_k<<<(tot + 255) / 256, 256, 0, stream>>>(W2, W2T, FEAT, ATT, tot); }
  { int tot = FEAT * ATT;
    wtrans_k<<<(tot + 255) / 256, 256, 0, stream>>>(lin_W, linWT, FEAT, ATT, tot); }

  // ---- layer 1 ----
  node_gemm_k<<<192, 128, 0, stream>>>(emb, EMB, W1T, WhT);
  el_er_k<<<(NHEAD * NN + 255) / 256, 256, 0, stream>>>(WhT, av1, el, er);
  row_stats_k<<<(NHEAD * NN) / 8, 256, 0, stream>>>(el, er, adjbits, mrow, srow);
  agg_k<<<192, 128, 0, stream>>>(WhT, el, er, mrow, srow, adjbits, at1);

  // ---- layer 2 ----
  node_gemm_k<<<192, 128, 0, stream>>>(at1, FEAT, W2T, WhT);
  el_er_k<<<(NHEAD * NN + 255) / 256, 256, 0, stream>>>(WhT, av2, el, er);
  row_stats_k<<<(NHEAD * NN) / 8, 256, 0, stream>>>(el, er, adjbits, mrow, srow);
  agg_k<<<192, 128, 0, stream>>>(WhT, el, er, mrow, srow, adjbits, at2);

  // ---- mix + fused pair scorer ----
  mix_k<<<(NN * FEAT + 255) / 256, 256, 0, stream>>>(at1, at2, a_mix, b_mix, atten);
  pair_k<<<NPAIRS / 16 / 4, 128, 0, stream>>>(atten, a_idx, b_idx, linWT, lin_b,
                                              emb_b, out_W, out_b, out);
}
